// GemNetOCBackbone_55009941128038
// MI455X (gfx1250) — matrix operand order repaired and checked by reading, not compiled, not run on hardware
//
#include <hip/hip_runtime.h>
#include <cstddef>

// ---------------------------------------------------------------------------
// GemNet-OC backbone for MI455X (gfx1250, wave32, WMMA bf16 -> f32 acc)
// ---------------------------------------------------------------------------
#define DA 256
#define DE 512
#define DT 64
#define DC 16
#define NR 128
#define NS 7
#define NB 2
#define INV_SQRT2 0.7071067811865476f

typedef __attribute__((ext_vector_type(16))) __bf16 v16bf;
typedef __attribute__((ext_vector_type(8)))  __bf16 v8bf;
typedef __attribute__((ext_vector_type(8)))  float  v8f;

__device__ __forceinline__ float silu_f(float x) {
  return x / (1.0f + __expf(-x));
}

__device__ __forceinline__ void atom_add_f32(float* p, float v) {
  unsafeAtomicAdd(p, v);   // hardware global_atomic_add_f32
}

// ---------------------------------------------------------------------------
// Generic WMMA GEMM:  C = epilogue(A (.* Amul) @ B)
//   A: [M,K] f32 row-major (optionally elementwise-multiplied by Amul [M,K])
//   B: [K,N] f32 row-major
//   epilogue: +bias[n], optional SiLU, optional +skip[m,n], *scale,
//             optional atomic scatter-add into C[scat[m]*N + n]
// Tiling: BM=128, BN=128, BK=32; 256 threads = 8 wave32 waves (4x2 grid),
// each wave owns a 32x64 tile = 2x4 v_wmma_f32_16x16x32_bf16 accumulators.
// Double-buffered LDS, software-pipelined: global loads for tile i+1 issue
// before the WMMAs of tile i; cvt+ds_store after; one barrier per k-iter.
// B staging: each thread owns a 4(k)x4(n) micro-tile, transposes it in
// registers and writes columns as single ds_store_b64 (no scalar b16 stores).
// ---------------------------------------------------------------------------
#define BM 128
#define BN 128
#define BK 32
#define LDSK (BK + 8)   // pad: 40 bf16 = 80B rows (16B aligned, bank-spread)

__global__ __launch_bounds__(256)
void gemm_wmma_kernel(const float* __restrict__ A,
                      const float* __restrict__ Amul,
                      const float* __restrict__ B,
                      const float* __restrict__ bias,
                      const float* __restrict__ skip,
                      const int*   __restrict__ scat,
                      float* __restrict__ C,
                      int M, int N, int K, int act, float scale)
{
  __shared__ alignas(16) __bf16 As[2][BM][LDSK];
  __shared__ alignas(16) __bf16 Bs[2][BN][LDSK];   // transposed: Bs[.][n][k]

  const int tid  = threadIdx.x;
  const int lane = tid & 31;
  const int wave = tid >> 5;      // 0..7  (wave32)
  const int wr   = wave >> 1;     // 0..3 -> 32-row group
  const int wc   = wave & 1;      // 0..1 -> 64-col group
  const int ln   = lane & 15;
  const int hi   = lane >> 4;     // 0/1 half-wave

  const int m0 = blockIdx.y * BM;
  const int n0 = blockIdx.x * BN;

  // A staging: 4 float4 chunks per thread (row-major, 4 rows spread by 32)
  int arow[4], ak[4];
  for (int i = 0; i < 4; ++i) {
    int c = tid + i * 256;          // 0..1023
    arow[i] = c >> 3;               // A tile row 0..127
    ak[i]   = (c & 7) << 2;         // A k within tile, step 4
  }
  // B staging: 4(k) x 4(n) micro-tile per thread; within a wave, lanes sweep
  // consecutive 16B chunks of one B row -> fully coalesced global loads.
  const int bk4 = wave * 4;         // k row base (0..28)
  const int bn4 = lane * 4;         // n col base (0..124)

  const bool kfull = (K & (BK - 1)) == 0;
  const bool aint  = kfull && (m0 + BM <= M);   // A tile fully interior
  const bool bint  = kfull && (n0 + BN <= N);   // B tile fully interior

  v8f zero8;
  for (int i = 0; i < 8; ++i) zero8[i] = 0.0f;
  v8f acc[2][4];
  for (int tr = 0; tr < 2; ++tr)
    for (int tc = 0; tc < 4; ++tc) acc[tr][tc] = zero8;

  float4 va[4];
  float  vbf[4][4];   // [k][n] micro-tile

  // ---- global -> register stage for the k-tile starting at kk ----
  auto stage_load = [&](int kk) {
    if (aint) {
      for (int i = 0; i < 4; ++i)
        va[i] = *(const float4*)(A + (size_t)(m0 + arow[i]) * K + kk + ak[i]);
      if (Amul) {
        for (int i = 0; i < 4; ++i) {
          float4 u = *(const float4*)(Amul + (size_t)(m0 + arow[i]) * K + kk + ak[i]);
          va[i].x *= u.x; va[i].y *= u.y; va[i].z *= u.z; va[i].w *= u.w;
        }
      }
    } else {
      for (int i = 0; i < 4; ++i) {
        int gm = m0 + arow[i], gk = kk + ak[i];
        float4 v = make_float4(0.f, 0.f, 0.f, 0.f);
        if (gm < M && gk < K) {
          v = *(const float4*)(A + (size_t)gm * K + gk);
          if (Amul) {
            float4 u = *(const float4*)(Amul + (size_t)gm * K + gk);
            v.x *= u.x; v.y *= u.y; v.z *= u.z; v.w *= u.w;
          }
        }
        va[i] = v;
      }
    }
    if (bint) {
      for (int j = 0; j < 4; ++j) {
        float4 v = *(const float4*)(B + (size_t)(kk + bk4 + j) * N + n0 + bn4);
        vbf[j][0] = v.x; vbf[j][1] = v.y; vbf[j][2] = v.z; vbf[j][3] = v.w;
      }
    } else {
      for (int j = 0; j < 4; ++j) {
        int gk = kk + bk4 + j, gn = n0 + bn4;
        float4 v = make_float4(0.f, 0.f, 0.f, 0.f);
        if (gk < K && gn < N)
          v = *(const float4*)(B + (size_t)gk * N + gn);
        vbf[j][0] = v.x; vbf[j][1] = v.y; vbf[j][2] = v.z; vbf[j][3] = v.w;
      }
    }
  };

  // ---- register -> LDS (with f32 -> bf16 convert) ----
  auto stage_store = [&](int buf) {
    for (int i = 0; i < 4; ++i) {
      As[buf][arow[i]][ak[i] + 0] = (__bf16)va[i].x;
      As[buf][arow[i]][ak[i] + 1] = (__bf16)va[i].y;
      As[buf][arow[i]][ak[i] + 2] = (__bf16)va[i].z;
      As[buf][arow[i]][ak[i] + 3] = (__bf16)va[i].w;
    }
    // 4x4 register transpose: write each output column (fixed n) as 4
    // consecutive bf16 along k -> single 8B-aligned ds_store_b64.
    for (int j = 0; j < 4; ++j) {
      __bf16* dst = &Bs[buf][bn4 + j][bk4];
      dst[0] = (__bf16)vbf[0][j];
      dst[1] = (__bf16)vbf[1][j];
      dst[2] = (__bf16)vbf[2][j];
      dst[3] = (__bf16)vbf[3][j];
    }
  };

  // ---- fragment loads + 8 WMMAs on buffer `buf` ----
  auto compute = [&](int buf) {
    // A 16x32 bf16: lane holds row m=ln; elem e -> k = 8*hi + 16*(e>=8) + (e&7)
    v16bf afrag[2];
    for (int tr = 0; tr < 2; ++tr) {
      const __bf16* p = &As[buf][wr * 32 + tr * 16 + ln][hi * 8];
      v8bf lo = *(const v8bf*)p;
      v8bf hg = *(const v8bf*)(p + 16);
      v16bf a;
      for (int e = 0; e < 8; ++e) { a[e] = lo[e]; a[e + 8] = hg[e]; }
      afrag[tr] = a;
    }
    // B 32x16 bf16: lane holds col n=ln; elem e -> k = 8*hi + 16*(e>=8) + (e&7)
    v16bf bfrag[4];
    for (int tc = 0; tc < 4; ++tc) {
      const __bf16* p = &Bs[buf][wc * 64 + tc * 16 + ln][hi * 8];
      v8bf p0 = *(const v8bf*)p;
      v8bf p1 = *(const v8bf*)(p + 16);
      v16bf b;
      for (int e = 0; e < 8; ++e) { b[e] = p0[e]; b[e + 8] = p1[e]; }
      bfrag[tc] = b;
    }
    for (int tr = 0; tr < 2; ++tr)
      for (int tc = 0; tc < 4; ++tc)
        acc[tr][tc] = __builtin_amdgcn_wmma_f32_16x16x32_bf16(
            false, afrag[tr], false, bfrag[tc], (short)0, acc[tr][tc],
            false, false);
  };

  const int nk = (K + BK - 1) / BK;

  // prologue: fill buffer 0
  stage_load(0);
  stage_store(0);
  __syncthreads();

  for (int it = 0; it < nk; ++it) {
    const int buf = it & 1;
    if (it + 2 < nk) {   // L2 warm-up two tiles ahead (global_prefetch_b8)
      __builtin_prefetch(A + (size_t)(m0 + arow[0]) * K + (it + 2) * BK, 0, 1);
      __builtin_prefetch(B + (size_t)((it + 2) * BK + bk4) * N + n0 + bn4, 0, 1);
    }
    if (it + 1 < nk) stage_load((it + 1) * BK);  // VMEM issues early
    compute(buf);                                // WMMAs hide VMEM latency
    if (it + 1 < nk) stage_store(buf ^ 1);       // cvt + ds_store after
    __syncthreads();
  }

  // ---- epilogue: C/D layout -> m = r + 8*hi, n = ln within each 16x16 ----
  for (int tr = 0; tr < 2; ++tr) {
    int gm_base = m0 + wr * 32 + tr * 16 + hi * 8;
    for (int tc = 0; tc < 4; ++tc) {
      int gn = n0 + wc * 64 + tc * 16 + ln;
      if (gn >= N) continue;
      float bv = bias ? bias[gn] : 0.0f;
      for (int r = 0; r < 8; ++r) {
        int gm = gm_base + r;
        if (gm >= M) continue;
        float v = acc[tr][tc][r] + bv;
        if (act) v = silu_f(v);
        if (skip) v += skip[(size_t)gm * N + gn];
        v *= scale;
        if (scat) atom_add_f32(&C[(size_t)scat[gm] * N + gn], v);
        else      C[(size_t)gm * N + gn] = v;
      }
    }
  }
}

// ---------------------------------------------------------------------------
// Elementwise / gather / scatter kernels
// ---------------------------------------------------------------------------
__global__ void zero_kernel(float* __restrict__ p, long n) {
  long i = (long)blockIdx.x * 256 + threadIdx.x;
  if (i < n) p[i] = 0.0f;
}

// Gaussian RBF with polynomial envelope (p=5). One thread per (edge, radial).
__global__ void rbf_kernel(const float* __restrict__ d, float* __restrict__ rbf, int E) {
  long idx = (long)blockIdx.x * 256 + threadIdx.x;
  if (idx >= (long)E * NR) return;
  int e = (int)(idx >> 7);
  int r = (int)(idx & 127);
  float dist = d[e];
  float u  = dist * (1.0f / 6.0f);
  float u2 = u * u;
  float u5 = u2 * u2 * u;
  float env = 1.0f - 21.0f * u5 + 35.0f * u5 * u - 15.0f * u5 * u2;
  env = (u < 1.0f) ? env : 0.0f;
  float mu = (float)r * (6.0f / 127.0f);
  const float gamma = 0.5f / ((6.0f / 128.0f) * (6.0f / 128.0f));
  float x = dist - mu;
  rbf[idx] = env * __expf(-gamma * x * x);
}

__global__ void init_h_kernel(const int* __restrict__ z,
                              const float* __restrict__ table,
                              float* __restrict__ h, int N) {
  long idx = (long)blockIdx.x * 256 + threadIdx.x;
  if (idx >= (long)N * DA) return;
  int n = (int)(idx >> 8);
  int d = (int)(idx & 255);
  h[idx] = table[(size_t)z[n] * DA + d];
}

// cat[e] = [h[src[e]](256) | h[tgt[e]](256) | rbf[e](128)]
__global__ void concat_kernel(const float* __restrict__ h,
                              const float* __restrict__ rbf,
                              const int* __restrict__ src,
                              const int* __restrict__ tgt,
                              float* __restrict__ cat, int E) {
  long idx = (long)blockIdx.x * 256 + threadIdx.x;
  if (idx >= (long)E * 640) return;
  int e = (int)(idx / 640);
  int j = (int)(idx - (long)e * 640);
  float v;
  if (j < 256)       v = h[(size_t)src[e] * DA + j];
  else if (j < 512)  v = h[(size_t)tgt[e] * DA + (j - 256)];
  else               v = rbf[(size_t)e * NR + (j - 512)];
  cat[idx] = v;
}

// cbf_emb[t,c] = rbf3[id3_ba[t],c] * sum_s T_s(cos) * W_sph[s,c]
// cos(s*theta) with theta=acos(c) == Chebyshev T_s(c) (exact).
__global__ void cbf_emb_kernel(const float* __restrict__ cosang,
                               const int* __restrict__ id3_ba,
                               const float* __restrict__ rbf3,
                               const float* __restrict__ W_sph,
                               float* __restrict__ cbf_emb, int T) {
  __shared__ float wsph[NS * DC];
  if (threadIdx.x < NS * DC) wsph[threadIdx.x] = W_sph[threadIdx.x];
  __syncthreads();
  long t = (long)blockIdx.x * 256 + threadIdx.x;
  if (t >= T) return;
  float c = cosang[t];
  c = fminf(1.0f, fmaxf(-1.0f, c));
  float cs[NS];
  cs[0] = 1.0f;
  cs[1] = c;
  for (int s = 2; s < NS; ++s) cs[s] = 2.0f * c * cs[s - 1] - cs[s - 2];
  const float* rb = rbf3 + (size_t)id3_ba[t] * DC;
  float* out = cbf_emb + (size_t)t * DC;
  for (int k = 0; k < DC; ++k) {
    float a = 0.0f;
    for (int s = 0; s < NS; ++s) a += cs[s] * wsph[s * DC + k];
    out[k] = rb[k] * a;
  }
}

// x3[id3_ca[t], :] += x_ba[id3_ba[t], :] * cbf_up[t, :]
__global__ void trip_scatter_kernel(const float* __restrict__ x_ba,
                                    const float* __restrict__ cbf_up,
                                    const int* __restrict__ id3_ba,
                                    const int* __restrict__ id3_ca,
                                    float* __restrict__ x3, int T) {
  long idx = (long)blockIdx.x * 256 + threadIdx.x;
  if (idx >= (long)T * DT) return;
  int t = (int)(idx >> 6);
  int c = (int)(idx & 63);
  int ba = id3_ba[t];
  int ca = id3_ca[t];
  float v = x_ba[(size_t)ba * DT + c] * cbf_up[(size_t)t * DT + c];
  atom_add_f32(&x3[(size_t)ca * DT + c], v);
}

// energy = sum_{n,d} h[n,d] * W_out[d]
__global__ void energy_kernel(const float* __restrict__ h,
                              const float* __restrict__ w,
                              float* __restrict__ out, long n) {
  __shared__ float sbuf[256];
  long i0 = (long)blockIdx.x * 256 + threadIdx.x;
  float v = 0.0f;
  for (long i = i0; i < n; i += (long)gridDim.x * 256)
    v += h[i] * w[i & 255];
  sbuf[threadIdx.x] = v;
  __syncthreads();
  for (int s = 128; s > 0; s >>= 1) {
    if (threadIdx.x < s) sbuf[threadIdx.x] += sbuf[threadIdx.x + s];
    __syncthreads();
  }
  if (threadIdx.x == 0) atom_add_f32(out, sbuf[0]);
}

// ---------------------------------------------------------------------------
// Host-side launch
// ---------------------------------------------------------------------------
static inline unsigned cdivu(long a, long b) { return (unsigned)((a + b - 1) / b); }

static inline void launch_gemm(hipStream_t s, const float* A, const float* Amul,
                               const float* B, const float* bias, const float* skip,
                               const int* scat, float* C, int M, int N, int K,
                               int act, float scale) {
  dim3 grid(cdivu(N, BN), cdivu(M, BM));
  gemm_wmma_kernel<<<grid, 256, 0, s>>>(A, Amul, B, bias, skip, scat, C,
                                        M, N, K, act, scale);
}

extern "C" void kernel_launch(void* const* d_in, const int* in_sizes, int n_in,
                              void* d_out, int out_size, void* d_ws, size_t ws_size,
                              hipStream_t stream) {
  const int*   atomic_numbers = (const int*)  d_in[0];
  const int*   edge_index     = (const int*)  d_in[1];
  const float* distances      = (const float*)d_in[2];
  const int*   id3_ba         = (const int*)  d_in[3];
  const int*   id3_ca         = (const int*)  d_in[4];
  const float* cos_angles     = (const float*)d_in[5];
  const float* atom_table     = (const float*)d_in[6];
  const float* W_edge         = (const float*)d_in[7];
  const float* b_edge         = (const float*)d_in[8];
  const float* W_rbf3         = (const float*)d_in[9];
  const float* W_sph          = (const float*)d_in[10];
  const float* W_cbf_up       = (const float*)d_in[11];
  const float* W_rbf_h        = (const float*)d_in[12];
  const float* W_ba           = (const float*)d_in[13];
  const float* b_ba           = (const float*)d_in[14];
  const float* W_down         = (const float*)d_in[15];
  const float* W_trip_up      = (const float*)d_in[16];
  const float* W_res_m        = (const float*)d_in[17];
  const float* b_res_m        = (const float*)d_in[18];
  const float* W_e2a          = (const float*)d_in[19];
  const float* W_res_h        = (const float*)d_in[20];
  const float* b_res_h        = (const float*)d_in[21];
  const float* W_out          = (const float*)d_in[22];
  (void)n_in; (void)out_size; (void)ws_size;

  const int N = in_sizes[0];
  const int E = in_sizes[2];
  const int T = in_sizes[3];
  const int* src = edge_index;       // edge_index[0, :]
  const int* tgt = edge_index + E;   // edge_index[1, :]

  // workspace layout (f32)
  float* ws = (float*)d_ws;
  size_t o = 0;
  float* rbf   = ws + o; o += (size_t)E * NR;
  float* rbfh  = ws + o; o += (size_t)E * DE;
  float* m     = ws + o; o += (size_t)E * DE;
  float* tmp1  = ws + o; o += (size_t)E * DE;
  float* x_ba  = ws + o; o += (size_t)E * DT;
  float* x3    = ws + o; o += (size_t)E * DT;
  float* rbf3  = ws + o; o += (size_t)E * DC;
  float* cemb  = ws + o; o += (size_t)T * DC;
  float* cup   = ws + o; o += (size_t)T * DT;
  float* h     = ws + o; o += (size_t)N * DA;
  float* htmp  = ws + o; o += (size_t)N * DA;
  float* cat   = ws + o; o += (size_t)E * (2 * DA + NR);

  // ---- shared basis ----
  rbf_kernel<<<cdivu((long)E * NR, 256), 256, 0, stream>>>(distances, rbf, E);
  launch_gemm(stream, rbf, nullptr, W_rbf3, nullptr, nullptr, nullptr,
              rbf3, E, DC, NR, 0, 1.0f);
  launch_gemm(stream, rbf, nullptr, W_rbf_h, nullptr, nullptr, nullptr,
              rbfh, E, DE, NR, 0, 1.0f);
  cbf_emb_kernel<<<cdivu(T, 256), 256, 0, stream>>>(cos_angles, id3_ba, rbf3,
                                                    W_sph, cemb, T);
  launch_gemm(stream, cemb, nullptr, W_cbf_up, nullptr, nullptr, nullptr,
              cup, T, DT, DC, 0, 1.0f);

  // ---- embeddings ----
  init_h_kernel<<<cdivu((long)N * DA, 256), 256, 0, stream>>>(atomic_numbers,
                                                              atom_table, h, N);
  concat_kernel<<<cdivu((long)E * 640, 256), 256, 0, stream>>>(h, rbf, src, tgt,
                                                               cat, E);
  launch_gemm(stream, cat, nullptr, W_edge, b_edge, nullptr, nullptr,
              m, E, DE, 2 * DA + NR, 1, 1.0f);

  // ---- interaction blocks ----
  for (int b = 0; b < NB; ++b) {
    // triplet message path
    launch_gemm(stream, m, nullptr, W_ba + (size_t)b * DE * DE,
                b_ba + (size_t)b * DE, nullptr, nullptr, tmp1, E, DE, DE, 1, 1.0f);
    launch_gemm(stream, tmp1, nullptr, W_down + (size_t)b * DE * DT,
                nullptr, nullptr, nullptr, x_ba, E, DT, DE, 0, 1.0f);
    zero_kernel<<<cdivu((long)E * DT, 256), 256, 0, stream>>>(x3, (long)E * DT);
    trip_scatter_kernel<<<cdivu((long)T * DT, 256), 256, 0, stream>>>(
        x_ba, cup, id3_ba, id3_ca, x3, T);
    launch_gemm(stream, x3, nullptr, W_trip_up + (size_t)b * DT * DE,
                nullptr, m, nullptr, m, E, DE, DT, 0, INV_SQRT2);
    // edge residual stack
    for (int r = 0; r < 2; ++r) {
      size_t wi = ((size_t)(b * 2 + r)) * 2;
      const float* W0 = W_res_m + (wi + 0) * DE * DE;
      const float* W1 = W_res_m + (wi + 1) * DE * DE;
      const float* c0 = b_res_m + (wi + 0) * DE;
      const float* c1 = b_res_m + (wi + 1) * DE;
      launch_gemm(stream, m, nullptr, W0, c0, nullptr, nullptr, tmp1, E, DE, DE, 1, 1.0f);
      launch_gemm(stream, tmp1, nullptr, W1, c1, m, nullptr, m, E, DE, DE, 1, INV_SQRT2);
    }
    // atom update: (m * rbf_h) @ W_e2a, atomically scattered onto h[tgt]
    launch_gemm(stream, m, rbfh, W_e2a + (size_t)b * DE * DA,
                nullptr, nullptr, tgt, h, E, DA, DE, 0, 1.0f);
    // atom residual stack
    for (int r = 0; r < 2; ++r) {
      size_t wi = ((size_t)(b * 2 + r)) * 2;
      const float* W0 = W_res_h + (wi + 0) * DA * DA;
      const float* W1 = W_res_h + (wi + 1) * DA * DA;
      const float* c0 = b_res_h + (wi + 0) * DA;
      const float* c1 = b_res_h + (wi + 1) * DA;
      launch_gemm(stream, h, nullptr, W0, c0, nullptr, nullptr, htmp, N, DA, DA, 1, 1.0f);
      launch_gemm(stream, htmp, nullptr, W1, c1, h, nullptr, h, N, DA, DA, 1, INV_SQRT2);
    }
  }

  // ---- energy ----
  zero_kernel<<<1, 256, 0, stream>>>((float*)d_out, 1);
  energy_kernel<<<1024, 256, 0, stream>>>(h, W_out, (float*)d_out, (long)N * DA);
}
